// SimCLRLoss_26577257627898
// MI455X (gfx1250) — matrix operand order repaired and checked by reading, not compiled, not run on hardware
//
#include <hip/hip_runtime.h>

// ---------------------------------------------------------------------------
// SimCLR loss on MI455X (gfx1250), wave32 + WMMA bf16 + async LDS staging.
//
//   N = 8192 rows, D = 256, T = 0.1  (1/T = 10)
//   GEMM: 34.4 GFLOP, data: 8 MB fp32 (L2-resident) -> compute bound.
//   32 rows/wave (2 M-blocks per B chunk). B tiles (8 KB contiguous) are
//   DMA'd into a 4-slot LDS ring with GLOBAL_LOAD_ASYNC_TO_LDS_B128
//   (ASYNCcnt, 3 tiles in flight, zero VGPR cost); WMMA consumes via
//   ds_load_b128 whose latency is trivially hidden.
// ---------------------------------------------------------------------------

#define N_ROWS 8192
#define DIM    256
#define INV_T  10.0f
#define EXP2K  14.4269504088896340736f  // 10 * log2(e)
#define NSEG   8
#define NTILES (N_ROWS / NSEG / 16)     // 64 column tiles per segment
#define TILE_BYTES 8192                 // 16 rows x 256 K x 2 B (contiguous)

typedef __attribute__((ext_vector_type(16))) __bf16 bf16x16;
typedef __attribute__((ext_vector_type(8)))  __bf16 bf16x8;
typedef __attribute__((ext_vector_type(8)))  float  f32x8;

__device__ __forceinline__ float fast_exp10cm10(float c) {
  // exp(10*c - 10) == exp2(c*K - K), K = 10*log2(e); one v_fma + one v_exp_f32
  return __builtin_amdgcn_exp2f(fmaf(c, EXP2K, -EXP2K));
}

// ---------------------------------------------------------------------------
// Kernel 1: per-row L2 normalize; emit bf16 unit vectors + fp32 inverse norms.
// ---------------------------------------------------------------------------
__global__ __launch_bounds__(256) void simclr_normalize_kernel(
    const float* __restrict__ feats, __bf16* __restrict__ fnb,
    float* __restrict__ invn) {
  const int wave = threadIdx.x >> 5;
  const int lane = threadIdx.x & 31;
  const int row  = blockIdx.x * 8 + wave;

  const float* rp = feats + (size_t)row * DIM + lane * 8;
  float4 x0 = *(const float4*)rp;
  float4 x1 = *(const float4*)(rp + 4);

  float ss = x0.x * x0.x + x0.y * x0.y + x0.z * x0.z + x0.w * x0.w +
             x1.x * x1.x + x1.y * x1.y + x1.z * x1.z + x1.w * x1.w;
#pragma unroll
  for (int m = 1; m < 32; m <<= 1) ss += __shfl_xor(ss, m, 32);

  float nrm = fmaxf(sqrtf(ss), 1e-8f);
  float inv = 1.0f / nrm;

  bf16x8 o;
  o[0] = (__bf16)(x0.x * inv); o[1] = (__bf16)(x0.y * inv);
  o[2] = (__bf16)(x0.z * inv); o[3] = (__bf16)(x0.w * inv);
  o[4] = (__bf16)(x1.x * inv); o[5] = (__bf16)(x1.y * inv);
  o[6] = (__bf16)(x1.z * inv); o[7] = (__bf16)(x1.w * inv);
  *(bf16x8*)(fnb + (size_t)row * DIM + lane * 8) = o;

  if (lane == 0) invn[row] = inv;
}

// ---------------------------------------------------------------------------
// Kernel 2: WMMA Gram sweep + masked exp-sum, async-LDS B staging.
// One wave per (32-row block, 1024-col segment). A in VGPRs (2 x 8 x v16bf).
// ---------------------------------------------------------------------------
__global__ __launch_bounds__(32) void simclr_lse_kernel(
    const __bf16* __restrict__ fnb, float* __restrict__ partS) {
  __shared__ char ldstile[4 * TILE_BYTES];  // 32 KB ring, 4 tile slots

  const int lane   = threadIdx.x & 31;
  const int rowblk = blockIdx.x >> 3;       // 0..255, 32 rows each
  const int seg    = blockIdx.x & (NSEG - 1);
  const int row0   = rowblk * 32;
  const int half   = lane >> 4;
  const int l4     = lane & 15;

  // --- Preload A: two 16x256 bf16 blocks in WMMA 16-bit A layout ---
  // lanes 0-15: row M=l4, K chunks {0..7,16..23}; lanes 16-31: {8..15,24..31}
  bf16x16 a0[8], a1[8];
  {
    const __bf16* ap0 = fnb + (size_t)(row0 + l4) * DIM + half * 8;
    const __bf16* ap1 = fnb + (size_t)(row0 + 16 + l4) * DIM + half * 8;
#pragma unroll
    for (int kc = 0; kc < 8; ++kc) {
      union { bf16x16 v; bf16x8 h[2]; } u;
      u.h[0] = *(const bf16x8*)(ap0 + kc * 32);
      u.h[1] = *(const bf16x8*)(ap0 + kc * 32 + 16);
      a0[kc] = u.v;
      u.h[0] = *(const bf16x8*)(ap1 + kc * 32);
      u.h[1] = *(const bf16x8*)(ap1 + kc * 32 + 16);
      a1[kc] = u.v;
    }
  }

  float racc0[8], racc1[8];
#pragma unroll
  for (int m = 0; m < 8; ++m) { racc0[m] = 0.0f; racc1[m] = 0.0f; }

  const int colbeg = seg * (N_ROWS / NSEG);
  const uint32_t ldsb = (uint32_t)(uintptr_t)(&ldstile[0]);

  // Async-DMA one 8 KB contiguous B tile into LDS slot: 16 async B128 ops
  // (32 lanes x 16 B each), tracked by ASYNCcnt, no VGPRs consumed.
  auto asyncTile = [&](int slot, int tileIdx) {
    uint64_t ga = (uint64_t)(uintptr_t)(fnb +
                  (size_t)(colbeg + tileIdx * 16) * DIM) + (uint32_t)(lane * 16);
    uint32_t la = ldsb + slot * TILE_BYTES + lane * 16;
#pragma unroll
    for (int i = 0; i < 16; ++i) {
      asm volatile("global_load_async_to_lds_b128 %0, %1, off offset:%2"
                   :: "v"(la), "v"(ga), "i"(i * 512)
                   : "memory");
    }
  };

  // Prologue: 3 tiles in flight (48 async ops).
  asyncTile(0, 0);
  asyncTile(1, 1 < NTILES ? 1 : 0);
  asyncTile(2, 2 < NTILES ? 2 : 0);

  for (int ct = 0; ct < NTILES; ++ct) {
    const int col0 = colbeg + ct * 16;

    // Keep the pipe full: issue tile ct+3 (clamped; duplicates are unused).
    {
      int t = ct + 3 < NTILES ? ct + 3 : NTILES - 1;
      asyncTile((ct + 3) & 3, t);
    }
    // Async loads retire in order: <=48 outstanding => tile ct's 16 landed.
    asm volatile("s_wait_asynccnt 48" ::: "memory");

    const char* lbuf = ldstile + (ct & 3) * TILE_BYTES;
    // B layout chunk kc for this lane: row l4 of tile, bytes
    // l4*512 + half*16 + kc*64 (+32 for second half). All ds_load_b128.
    auto ldsChunk = [&](bf16x16& dst, int kc) {
      const char* p = lbuf + l4 * 512 + half * 16 + kc * 64;
      union { bf16x16 v; bf16x8 h[2]; } u;
      u.h[0] = *(const bf16x8*)(p);
      u.h[1] = *(const bf16x8*)(p + 32);
      dst = u.v;
    };

    f32x8 c0 = {}, c1 = {};
    bf16x16 bc[2];
    ldsChunk(bc[0], 0);
#pragma unroll
    for (int kc = 0; kc < 8; ++kc) {
      if (kc < 7) ldsChunk(bc[(kc + 1) & 1], kc + 1);
      c0 = __builtin_amdgcn_wmma_f32_16x16x32_bf16(
          false, a0[kc], false, bc[kc & 1], (short)0, c0, false, false);
      c1 = __builtin_amdgcn_wmma_f32_16x16x32_bf16(
          false, a1[kc], false, bc[kc & 1], (short)0, c1, false, false);
    }

    const int mycol = col0 + l4;
    // Diagonal appears only in the two tiles aligned with our row blocks.
    if (col0 == row0 || col0 == row0 + 16) {  // uniform scalar branch
#pragma unroll
      for (int m = 0; m < 8; ++m) {
        const int g0 = row0 + m + half * 8;
        const int g1 = row0 + 16 + m + half * 8;
        float e0 = fast_exp10cm10(c0[m]);
        float e1 = fast_exp10cm10(c1[m]);
        racc0[m] += (g0 == mycol) ? 0.0f : e0;
        racc1[m] += (g1 == mycol) ? 0.0f : e1;
      }
    } else {
#pragma unroll
      for (int m = 0; m < 8; ++m) {
        racc0[m] += fast_exp10cm10(c0[m]);
        racc1[m] += fast_exp10cm10(c1[m]);
      }
    }
  }

  // Reduce each row across its 16 lanes (lane bit 4 untouched).
#pragma unroll
  for (int m = 0; m < 8; ++m) {
    float s0 = racc0[m], s1 = racc1[m];
    s0 += __shfl_xor(s0, 1, 32); s1 += __shfl_xor(s1, 1, 32);
    s0 += __shfl_xor(s0, 2, 32); s1 += __shfl_xor(s1, 2, 32);
    s0 += __shfl_xor(s0, 4, 32); s1 += __shfl_xor(s1, 4, 32);
    s0 += __shfl_xor(s0, 8, 32); s1 += __shfl_xor(s1, 8, 32);
    racc0[m] = s0; racc1[m] = s1;
  }
  if (l4 == 0) {
    float* base = partS + (size_t)seg * N_ROWS;
#pragma unroll
    for (int m = 0; m < 8; ++m) {
      base[row0 + m + half * 8]      = racc0[m];
      base[row0 + 16 + m + half * 8] = racc1[m];
    }
  }
}

// ---------------------------------------------------------------------------
// Kernel 3: exact fp32 positive logit + per-row loss.
// ---------------------------------------------------------------------------
__global__ __launch_bounds__(256) void simclr_pos_kernel(
    const float* __restrict__ feats, const float* __restrict__ invn,
    const float* __restrict__ partS, float* __restrict__ rowloss) {
  const int wave = threadIdx.x >> 5;
  const int lane = threadIdx.x & 31;
  const int row  = blockIdx.x * 8 + wave;
  const int prow = row ^ (N_ROWS / 2);  // (i +/- n/2) mod n

  const float* rp = feats + (size_t)row * DIM + lane * 8;
  const float* pp = feats + (size_t)prow * DIM + lane * 8;
  float4 a0 = *(const float4*)rp, a1 = *(const float4*)(rp + 4);
  float4 b0 = *(const float4*)pp, b1 = *(const float4*)(pp + 4);

  float dot = a0.x * b0.x + a0.y * b0.y + a0.z * b0.z + a0.w * b0.w +
              a1.x * b1.x + a1.y * b1.y + a1.z * b1.z + a1.w * b1.w;
#pragma unroll
  for (int m = 1; m < 32; m <<= 1) dot += __shfl_xor(dot, m, 32);

  if (lane == 0) {
    float S = 0.0f;
#pragma unroll
    for (int s = 0; s < NSEG; ++s) S += partS[(size_t)s * N_ROWS + row];
    float lse = __logf(S) + INV_T;
    float pos = INV_T * dot * invn[row] * invn[prow];
    rowloss[row] = lse - pos;
  }
}

// ---------------------------------------------------------------------------
// Kernel 4: fixed-order mean reduction (deterministic; no fp32 atomics).
// ---------------------------------------------------------------------------
__global__ __launch_bounds__(256) void simclr_mean_kernel(
    const float* __restrict__ rowloss, float* __restrict__ out) {
  __shared__ float sm[256];
  float s = 0.0f;
  for (int i = threadIdx.x; i < N_ROWS; i += 256) s += rowloss[i];
  sm[threadIdx.x] = s;
  __syncthreads();
  for (int off = 128; off > 0; off >>= 1) {
    if (threadIdx.x < off) sm[threadIdx.x] += sm[threadIdx.x + off];
    __syncthreads();
  }
  if (threadIdx.x == 0) out[0] = sm[0] * (1.0f / (float)N_ROWS);
}

// ---------------------------------------------------------------------------
// Workspace layout (bytes):
//   [0, 4 MiB)      : bf16 normalized features  (8192*256*2)
//   next 32 KiB     : fp32 inverse norms        (8192*4)
//   next 256 KiB    : fp32 segment partials     (8*8192*4)
//   next 32 KiB     : fp32 per-row loss         (8192*4)
// ---------------------------------------------------------------------------
extern "C" void kernel_launch(void* const* d_in, const int* in_sizes, int n_in,
                              void* d_out, int out_size, void* d_ws,
                              size_t ws_size, hipStream_t stream) {
  (void)in_sizes; (void)n_in; (void)out_size; (void)ws_size;
  const float* feats = (const float*)d_in[0];
  float* out = (float*)d_out;

  char* ws = (char*)d_ws;
  __bf16* fnb    = (__bf16*)(ws);
  float* invn    = (float*)(ws + (size_t)N_ROWS * DIM * 2);
  float* partS   = (float*)(ws + (size_t)N_ROWS * DIM * 2 + N_ROWS * 4);
  float* rowloss = (float*)(ws + (size_t)N_ROWS * DIM * 2 + N_ROWS * 4 +
                            (size_t)NSEG * N_ROWS * 4);

  simclr_normalize_kernel<<<dim3(N_ROWS / 8), dim3(256), 0, stream>>>(
      feats, fnb, invn);
  simclr_lse_kernel<<<dim3((N_ROWS / 32) * NSEG), dim3(32), 0, stream>>>(
      fnb, partS);
  simclr_pos_kernel<<<dim3(N_ROWS / 8), dim3(256), 0, stream>>>(
      feats, invn, partS, rowloss);
  simclr_mean_kernel<<<dim3(1), dim3(256), 0, stream>>>(rowloss, out);
}
